// LRNet_CIFAR10_ver2XXX_36009005810029
// MI455X (gfx1250) — matrix operand order repaired and checked by reading, not compiled, not run on hardware
//
#include <hip/hip_runtime.h>

typedef __attribute__((ext_vector_type(16))) _Float16 v16h;
typedef __attribute__((ext_vector_type(8)))  float    v8f;
typedef __attribute__((ext_vector_type(4)))  unsigned v4u;
typedef __attribute__((ext_vector_type(8)))  int      v8i;
typedef __attribute__((ext_vector_type(4)))  int      v4i;

#define EPSF 1e-5f

// ---------------------------------------------------------------------------
// CDNA5 async global->LDS copy (ASYNCcnt path). Generic LDS pointer low 32
// bits == wave-relative LDS byte offset (ISA: LDS aperture uses addr[31:0]).
// ---------------------------------------------------------------------------
__device__ __forceinline__ void async_b128(void* lds_dst, const void* gsrc) {
    unsigned lds_off = (unsigned)(unsigned long long)lds_dst;
    asm volatile("global_load_async_to_lds_b128 %0, %1, off"
                 :: "v"(lds_off), "v"((unsigned long long)gsrc)
                 : "memory");
}
__device__ __forceinline__ void wait_async0() {
    asm volatile("s_wait_asynccnt 0x0" ::: "memory");
}

// ---------------------------------------------------------------------------
// Tensor Data Mover: DMA a 2D tile (64 rows x 32 halves, row stride strideK
// elements) from global into LDS, HW-padding each 64B row by 48B so LDS row
// stride is 112B (56 halves) -- matches the compute tiles' padded layout.
// D# packing per CDNA5 ISA 8.3/8.4. Issued once per wave (EXEC-ignored op);
// caller gates to wave 0 and waits TENSORcnt==0 before the block barrier.
// ---------------------------------------------------------------------------
__device__ __forceinline__ void tdm_load_tile(void* lds_dst, const void* gsrc,
                                              int remK, int remRows, int strideK) {
    unsigned long long ga = (unsigned long long)gsrc;
    v4u g0;
    g0[0] = 1u;                                         // count=1 (valid user D#)
    g0[1] = (unsigned)(unsigned long long)lds_dst;      // lds_addr (bytes)
    g0[2] = (unsigned)ga;                               // global_addr[31:0]
    g0[3] = (unsigned)((ga >> 32) & 0x1FFFFFFull)       // global_addr[56:32]
          | (2u << 30);                                 // type=2 ("image")
    v8i g1;
    g1[0] = (int)((1u << 16)        // data_size = 1 -> 2 bytes
                | (1u << 20)        // pad_enable
                | (3u << 22)        // pad_interval = 16 DWORDs (one 64B row)
                | (11u << 25));     // pad_amount   = 12 DWORDs (48B)
    g1[1] = (int)(((unsigned)remK & 0xFFFFu) << 16);                       // tensor_dim0 lo
    g1[2] = (int)((((unsigned)remK >> 16) & 0xFFFFu)
                | (((unsigned)remRows & 0xFFFFu) << 16));                  // dim0 hi | dim1 lo
    g1[3] = (int)((((unsigned)remRows >> 16) & 0xFFFFu) | (32u << 16));    // dim1 hi | tile_dim0=32
    g1[4] = 64;                                                            // tile_dim1=64, tile_dim2=0
    g1[5] = strideK;                                                       // tensor_dim0_stride lo
    g1[6] = 0;
    g1[7] = 0;
    v4i gz4 = {0, 0, 0, 0};
    v8i gz8 = {0, 0, 0, 0, 0, 0, 0, 0};
    __builtin_amdgcn_tensor_load_to_lds(g0, g1, gz4, gz4, gz8, 0);  // 6-arg (clang-23)
}
__device__ __forceinline__ void wait_tensor0() {
    __builtin_amdgcn_s_wait_tensorcnt(0);
}

// ---------------------------------------------------------------------------
// Weight prep: ternary distribution -> (w_mean, w_var) in f16
// ---------------------------------------------------------------------------
__global__ void prep_ternary(const float* __restrict__ a, const float* __restrict__ b,
                             _Float16* __restrict__ wm, _Float16* __restrict__ wv, int n) {
    int i = blockIdx.x * 256 + threadIdx.x;
    if (i >= n) return;
    float p0 = 1.f / (1.f + expf(-a[i]));
    float p1 = (1.f - p0) * (1.f / (1.f + expf(-b[i])));
    float m = 2.f * p1 + p0 - 1.f;
    float v = (1.f - p0) - m * m;
    wm[i] = (_Float16)m;
    wv[i] = (_Float16)v;
}

__global__ void prep_f16(const float* __restrict__ src, _Float16* __restrict__ dst, int n) {
    int i = blockIdx.x * 256 + threadIdx.x;
    if (i < n) dst[i] = (_Float16)src[i];
}

// fc2 [10][1024] -> padded [64][1024] f16 (rows 10..63 zero, keeps async loads in-bounds)
__global__ void prep_fc2(const float* __restrict__ src, _Float16* __restrict__ dst) {
    int i = blockIdx.x * 256 + threadIdx.x;   // 64*1024
    int m = i >> 10, k = i & 1023;
    dst[i] = (m < 10) ? (_Float16)src[m * 1024 + k] : (_Float16)0.f;
}

// x: NCHW f32 [256,3,32,32] -> channel-major f16 [3][256*32*32]
__global__ void prep_x(const float* __restrict__ src, _Float16* __restrict__ dst) {
    int i = blockIdx.x * 256 + threadIdx.x;   // 3*262144
    int ci = i >> 18;
    int n  = i & 262143;
    int b  = n >> 10;
    int yx = n & 1023;
    dst[i] = (_Float16)src[((b * 3 + ci) << 10) + yx];
}

// ---------------------------------------------------------------------------
// Implicit-GEMM LR-conv, fused mean+variance paths.
// Block: 256 threads = 8 wave32, tile 64(M) x 64(N); each wave: 16x32 strip
// (one A-frag pair -> 4 WMMAs per K=32 step). Double-buffered LDS; weight
// tiles staged by the Tensor Data Mover (WMODE==1) or guarded scalar loads
// (WMODE==0, layer 1's K=27).
// EPI 0: erf(m*rsqrt(2(v+eps))) -> f16 [CO][N]
// EPI 1: m + sqrt(v+eps)*noise  -> f32 [CO][N]  (layer 6, HW==16)
// ---------------------------------------------------------------------------
template <int CI, int CO, int HIN, int WIN, int STRIDE, int HOUT, int WOUT, int EPI, int WMODE>
__global__ __launch_bounds__(256)
void conv_lr_wmma(const _Float16* __restrict__ in,      // [CI][256*HIN*WIN]
                  const _Float16* __restrict__ wmean,   // [CO][CI*9]
                  const _Float16* __restrict__ wvar,    // [CO][CI*9]
                  const float*    __restrict__ bias,    // [CO]
                  const float*    __restrict__ noise,   // NCHW [256,512,4,4] (EPI==1)
                  _Float16*       __restrict__ out16,   // EPI==0
                  float*          __restrict__ outf)    // EPI==1
{
    constexpr int K  = CI * 9;
    constexpr int N  = 256 * HOUT * WOUT;
    constexpr int HW = HOUT * WOUT;

    __shared__ __attribute__((aligned(16))) _Float16 Am[2][64][56];
    __shared__ __attribute__((aligned(16))) _Float16 Av[2][64][56];
    __shared__ __attribute__((aligned(16))) _Float16 Bs[2][64][56];

    const int tid  = threadIdx.x;
    const int lane = tid & 31;
    const int wave = tid >> 5;
    const int wRow = wave >> 1;      // 0..3 (M)
    const int wCol = wave & 1;       // 0..1 (N, 32-wide strip)
    const int coBase = blockIdx.x * 64;
    const int nBase  = blockIdx.y * 64;

    // staging coords: 64 rows x 32 k, 8 halves (16B) per thread
    const int wr  = tid >> 2;
    const int wk0 = (tid & 3) * 8;
    const int bn  = tid >> 2;
    const int bk0 = (tid & 3) * 8;

    const int nIdx = nBase + bn;
    const int bb   = nIdx / HW;
    const int rem  = nIdx % HW;
    const int ho   = rem / WOUT;
    const int wo   = rem % WOUT;

    auto stageW = [&](int buf, int kb) {
        if (WMODE == 1) {
            if (wave == 0) {        // one TDM descriptor pair per block per slab
                tdm_load_tile(&Am[buf][0][0], &wmean[coBase * K + kb], K - kb, CO - coBase, K);
                tdm_load_tile(&Av[buf][0][0], &wvar [coBase * K + kb], K - kb, CO - coBase, K);
            }
        } else {
#pragma unroll
            for (int i = 0; i < 8; ++i) {
                int k = kb + wk0 + i;
                _Float16 am = (_Float16)0.f, av = (_Float16)0.f;
                if (k < K) {
                    am = wmean[(coBase + wr) * K + k];
                    av = wvar [(coBase + wr) * K + k];
                }
                Am[buf][wr][wk0 + i] = am;
                Av[buf][wr][wk0 + i] = av;
            }
        }
    };
    auto stageB = [&](int buf, int kb) {
#pragma unroll
        for (int i = 0; i < 8; ++i) {
            int k = kb + bk0 + i;
            _Float16 val = (_Float16)0.f;
            if (k < K) {
                int ci = k / 9;
                int r  = k - ci * 9;
                int kh = r / 3;
                int kw = r - kh * 3;
                int iy = ho * STRIDE + kh - 1;
                int ix = wo * STRIDE + kw - 1;
                if (iy >= 0 && iy < HIN && ix >= 0 && ix < WIN)
                    val = in[ci * (256 * HIN * WIN) + bb * (HIN * WIN) + iy * WIN + ix];
            }
            Bs[buf][bn][bk0 + i] = val;
        }
    };
    auto waitStage = [&]() {
        if (WMODE == 1) { if (wave == 0) wait_tensor0(); }
    };

    v8f accM[2] = {v8f{}, v8f{}};
    v8f accV[2] = {v8f{}, v8f{}};

    stageW(0, 0);
    stageB(0, 0);
    waitStage();
    __syncthreads();

    for (int kb = 0; kb < K; kb += 32) {
        const int cur = (kb >> 5) & 1;
        const int nxt = cur ^ 1;
        if (kb + 32 < K) {           // DMA next slab while computing this one
            stageW(nxt, kb + 32);
            stageB(nxt, kb + 32);
        }

        const int mrow = wRow * 16 + (lane & 15);
        const int kg   = lane >> 4;
        v16h aM, aV;
#pragma unroll
        for (int i = 0; i < 8; ++i) {
            aM[i]     = Am[cur][mrow][kg * 8 + i];
            aM[8 + i] = Am[cur][mrow][16 + kg * 8 + i];
            aV[i]     = Av[cur][mrow][kg * 8 + i];
            aV[8 + i] = Av[cur][mrow][16 + kg * 8 + i];
        }
        const int nr0 = wCol * 32 + (lane & 15);
        v16h b0, b1, q0, q1;
#pragma unroll
        for (int i = 0; i < 16; ++i) {
            _Float16 x0 = Bs[cur][nr0][kg * 8 + i + (i & 8)];
            _Float16 x1 = Bs[cur][nr0 + 16][kg * 8 + i + (i & 8)];
            b0[i] = x0; q0[i] = x0 * x0;
            b1[i] = x1; q1[i] = x1 * x1;
        }
        accM[0] = __builtin_amdgcn_wmma_f32_16x16x32_f16(false, aM, false, b0, (short)0, accM[0], false, false);
        accV[0] = __builtin_amdgcn_wmma_f32_16x16x32_f16(false, aV, false, q0, (short)0, accV[0], false, false);
        accM[1] = __builtin_amdgcn_wmma_f32_16x16x32_f16(false, aM, false, b1, (short)0, accM[1], false, false);
        accV[1] = __builtin_amdgcn_wmma_f32_16x16x32_f16(false, aV, false, q1, (short)0, accV[1], false, false);

        waitStage();
        __syncthreads();
    }

    // ---- fused epilogue ----
    const int mHalf = lane >> 4;
#pragma unroll
    for (int s = 0; s < 2; ++s) {
        const int nOut = nBase + wCol * 32 + s * 16 + (lane & 15);
#pragma unroll
        for (int r = 0; r < 8; ++r) {
            int co = coBase + wRow * 16 + mHalf * 8 + r;
            float mval = accM[s][r] + bias[co];
            float vval = accV[s][r];
            if (EPI == 0) {
                float sgn = erff(mval * rsqrtf(2.0f * (vval + EPSF)));
                out16[co * N + nOut] = (_Float16)sgn;
            } else {
                int b  = nOut >> 4;           // HW == 16
                int yx = nOut & 15;
                float z = mval + sqrtf(vval + EPSF) * noise[b * (512 * 16) + co * 16 + yx];
                outf[co * N + nOut] = z;
            }
        }
    }
}

// ---------------------------------------------------------------------------
// BatchNorm over (B,H,W) per channel on z [512][4096] f32
// ---------------------------------------------------------------------------
__global__ __launch_bounds__(256)
void bn_stats(const float* __restrict__ z, float* __restrict__ mu, float* __restrict__ rstd) {
    __shared__ float s1[256];
    __shared__ float s2[256];
    int c = blockIdx.x, tid = threadIdx.x;
    float a = 0.f, b = 0.f;
    for (int i = tid; i < 4096; i += 256) {
        float v = z[c * 4096 + i];
        a += v; b += v * v;
    }
    s1[tid] = a; s2[tid] = b;
    __syncthreads();
    for (int s = 128; s > 0; s >>= 1) {
        if (tid < s) { s1[tid] += s1[tid + s]; s2[tid] += s2[tid + s]; }
        __syncthreads();
    }
    if (tid == 0) {
        float m = s1[0] * (1.f / 4096.f);
        float v = s2[0] * (1.f / 4096.f) - m * m;
        mu[c] = m;
        rstd[c] = rsqrtf(v + EPSF);
    }
}

// normalize + relu, transpose to GEMM B-layout zn[f=c*16+yx][b]  (f16 [8192][256])
__global__ void bn_apply(const float* __restrict__ z, const float* __restrict__ mu,
                         const float* __restrict__ rstd, const float* __restrict__ gamma,
                         const float* __restrict__ beta, _Float16* __restrict__ zn) {
    int idx = blockIdx.x * 256 + threadIdx.x;    // 512*4096
    int c = idx >> 12;
    int n = idx & 4095;
    int b  = n >> 4;
    int yx = n & 15;
    float v = gamma[c] * (z[idx] - mu[c]) * rstd[c] + beta[c];
    v = fmaxf(v, 0.f);
    zn[(c * 16 + yx) * 256 + b] = (_Float16)v;
}

// ---------------------------------------------------------------------------
// WMMA GEMM: out = A[M][K] x B[K][N]; 64x64 tile, double-buffered,
// async(ASYNCcnt) A staging. Requires M%64==0, K%32==0 (caller pads A).
// EPI 0: relu(acc+bias) -> f16 [M][N]
// EPI 1: acc+bias       -> f32 [N][Mvalid]  (logits, store only m<Mvalid)
// ---------------------------------------------------------------------------
template <int EPI>
__global__ __launch_bounds__(256)
void gemm_wmma(int M, int K, int N, int Mvalid,
               const _Float16* __restrict__ A, const _Float16* __restrict__ B,
               const float* __restrict__ bias,
               _Float16* __restrict__ out16, float* __restrict__ outf)
{
    __shared__ __attribute__((aligned(16))) _Float16 As[2][64][56];
    __shared__ __attribute__((aligned(16))) _Float16 Bs[2][64][56];

    const int tid  = threadIdx.x;
    const int lane = tid & 31;
    const int wave = tid >> 5;
    const int wRow = wave >> 1, wCol = wave & 1;
    const int mBase = blockIdx.x * 64;
    const int nBase = blockIdx.y * 64;
    const int wr = tid >> 2, wk0 = (tid & 3) * 8;
    const int bn = tid >> 2, bk0 = (tid & 3) * 8;

    auto stageA = [&](int buf, int kb) {
        async_b128(&As[buf][wr][wk0], &A[(mBase + wr) * K + kb + wk0]);
    };
    auto stageB = [&](int buf, int kb) {
#pragma unroll
        for (int i = 0; i < 8; ++i) {
            int k = kb + bk0 + i;
            Bs[buf][bn][bk0 + i] = B[k * N + nBase + bn];
        }
    };

    v8f acc[2] = {v8f{}, v8f{}};

    stageA(0, 0);
    stageB(0, 0);
    wait_async0();
    __syncthreads();

    for (int kb = 0; kb < K; kb += 32) {
        const int cur = (kb >> 5) & 1;
        const int nxt = cur ^ 1;
        if (kb + 32 < K) {
            stageA(nxt, kb + 32);
            stageB(nxt, kb + 32);
        }

        const int mrow = wRow * 16 + (lane & 15);
        const int kg   = lane >> 4;
        v16h aF, b0, b1;
#pragma unroll
        for (int i = 0; i < 8; ++i) {
            aF[i]     = As[cur][mrow][kg * 8 + i];
            aF[8 + i] = As[cur][mrow][16 + kg * 8 + i];
        }
        const int nr0 = wCol * 32 + (lane & 15);
#pragma unroll
        for (int i = 0; i < 16; ++i) {
            b0[i] = Bs[cur][nr0][kg * 8 + i + (i & 8)];
            b1[i] = Bs[cur][nr0 + 16][kg * 8 + i + (i & 8)];
        }
        acc[0] = __builtin_amdgcn_wmma_f32_16x16x32_f16(false, aF, false, b0, (short)0, acc[0], false, false);
        acc[1] = __builtin_amdgcn_wmma_f32_16x16x32_f16(false, aF, false, b1, (short)0, acc[1], false, false);

        wait_async0();
        __syncthreads();
    }

    const int mHalf = lane >> 4;
#pragma unroll
    for (int s = 0; s < 2; ++s) {
        const int nOut = nBase + wCol * 32 + s * 16 + (lane & 15);
#pragma unroll
        for (int r = 0; r < 8; ++r) {
            int m = mBase + wRow * 16 + mHalf * 8 + r;
            if (m < Mvalid) {
                float v = acc[s][r] + bias[m];
                if (EPI == 0) out16[m * N + nOut] = (_Float16)fmaxf(v, 0.f);
                else          outf[nOut * Mvalid + m] = v;
            }
        }
    }
}

// ---------------------------------------------------------------------------
extern "C" void kernel_launch(void* const* d_in, const int* in_sizes, int n_in,
                              void* d_out, int out_size, void* d_ws, size_t ws_size,
                              hipStream_t stream) {
    (void)in_sizes; (void)n_in; (void)out_size; (void)ws_size;

    const float* x    = (const float*)d_in[0];
    const float* Aa[6] = {(const float*)d_in[1],  (const float*)d_in[4],  (const float*)d_in[7],
                          (const float*)d_in[10], (const float*)d_in[13], (const float*)d_in[16]};
    const float* Bb[6] = {(const float*)d_in[2],  (const float*)d_in[5],  (const float*)d_in[8],
                          (const float*)d_in[11], (const float*)d_in[14], (const float*)d_in[17]};
    const float* Cb[6] = {(const float*)d_in[3],  (const float*)d_in[6],  (const float*)d_in[9],
                          (const float*)d_in[12], (const float*)d_in[15], (const float*)d_in[18]};
    const float* bn_g  = (const float*)d_in[19];
    const float* bn_b  = (const float*)d_in[20];
    const float* fc1w  = (const float*)d_in[21];
    const float* fc1b  = (const float*)d_in[22];
    const float* fc2w  = (const float*)d_in[23];
    const float* fc2b  = (const float*)d_in[24];
    const float* noise = (const float*)d_in[25];

    char* ws = (char*)d_ws;
    size_t off = 0;
    auto alloc = [&](size_t bytes) -> char* {
        char* p = ws + off;
        off = (off + bytes + 255) & ~(size_t)255;
        return p;
    };

    _Float16* x16  = (_Float16*)alloc(3UL * 262144 * 2);       // 1.5 MB
    _Float16* actA = (_Float16*)alloc(128UL * 262144 * 2);     // 64 MB ping
    _Float16* actB = (_Float16*)alloc(128UL * 65536 * 2);      // 16 MB pong

    static const size_t wElems[6] = {3456, 147456, 294912, 589824, 1179648, 2359296};
    _Float16 *WM[6], *WV[6];
    for (int l = 0; l < 6; ++l) {
        WM[l] = (_Float16*)alloc(wElems[l] * 2);
        WV[l] = (_Float16*)alloc(wElems[l] * 2);
    }
    _Float16* fc1_16 = (_Float16*)alloc(1024UL * 8192 * 2);    // 16 MB
    _Float16* fc2_16 = (_Float16*)alloc(64UL * 1024 * 2);      // padded to 64 rows
    float* mu   = (float*)alloc(512 * 4);
    float* rstd = (float*)alloc(512 * 4);

    // region aliases (lifetimes disjoint)
    float*    zbuf = (float*)actB;       // layer-6 sample, 512*4096 f32 = 8 MB
    _Float16* zn   = (_Float16*)actA;    // BN output [8192][256] f16 = 4 MB
    _Float16* h1   = (_Float16*)actB;    // fc1 output [1024][256] f16 (z dead by then)

    // ---- weight / input prep ----
    for (int l = 0; l < 6; ++l) {
        int n = (int)wElems[l];
        prep_ternary<<<(n + 255) / 256, 256, 0, stream>>>(Aa[l], Bb[l], WM[l], WV[l], n);
    }
    prep_f16<<<8388608 / 256, 256, 0, stream>>>(fc1w, fc1_16, 8388608);
    prep_fc2<<<65536 / 256, 256, 0, stream>>>(fc2w, fc2_16);
    prep_x<<<786432 / 256, 256, 0, stream>>>(x, x16);

    // ---- LR-conv stack ----
    conv_lr_wmma<3,  128, 32, 32, 1, 32, 32, 0, 0><<<dim3(2, 4096), 256, 0, stream>>>(x16,  WM[0], WV[0], Cb[0], nullptr, actA, nullptr);
    conv_lr_wmma<128,128, 32, 32, 2, 16, 16, 0, 1><<<dim3(2, 1024), 256, 0, stream>>>(actA, WM[1], WV[1], Cb[1], nullptr, actB, nullptr);
    conv_lr_wmma<128,256, 16, 16, 1, 16, 16, 0, 1><<<dim3(4, 1024), 256, 0, stream>>>(actB, WM[2], WV[2], Cb[2], nullptr, actA, nullptr);
    conv_lr_wmma<256,256, 16, 16, 2,  8,  8, 0, 1><<<dim3(4,  256), 256, 0, stream>>>(actA, WM[3], WV[3], Cb[3], nullptr, actB, nullptr);
    conv_lr_wmma<256,512,  8,  8, 1,  8,  8, 0, 1><<<dim3(8,  256), 256, 0, stream>>>(actB, WM[4], WV[4], Cb[4], nullptr, actA, nullptr);
    conv_lr_wmma<512,512,  8,  8, 2,  4,  4, 1, 1><<<dim3(8,   64), 256, 0, stream>>>(actA, WM[5], WV[5], Cb[5], noise, nullptr, zbuf);

    // ---- batchnorm + relu + transpose ----
    bn_stats<<<512, 256, 0, stream>>>(zbuf, mu, rstd);
    bn_apply<<<8192, 256, 0, stream>>>(zbuf, mu, rstd, bn_g, bn_b, zn);

    // ---- FC head ----
    gemm_wmma<0><<<dim3(16, 4), 256, 0, stream>>>(1024, 8192, 256, 1024, fc1_16, zn, fc1b, h1, nullptr);
    gemm_wmma<1><<<dim3(1, 4), 256, 0, stream>>>(64, 1024, 256, 10, fc2_16, h1, fc2b, nullptr, (float*)d_out);
}
